// MoE_53042846105633
// MI455X (gfx1250) — matrix operand order repaired and checked by reading, not compiled, not run on hardware
//
#include <hip/hip_runtime.h>
#include <hip/hip_bf16.h>

typedef __attribute__((ext_vector_type(16))) __bf16 v16bf;
typedef __attribute__((ext_vector_type(4)))  __bf16 v4bf;
typedef __attribute__((ext_vector_type(8)))  float  v8f;

#define DIN  3072
#define RHID 128
#define NEXP 8
#define EHID 2048
#define NCLS 10
#define NB   1024

#define TM  128  // sample tile (expert kernel)
#define THD 64   // hidden tile (expert kernel)
#define KC  32   // WMMA K step (bf16)
#define NITER (DIN / KC)   // 96 (even)

__device__ __forceinline__ uint2 pack4bf(float f0, float f1, float f2, float f3) {
    v4bf v;
    v[0] = (__bf16)f0; v[1] = (__bf16)f1;   // RNE converts -> packed cvt insts
    v[2] = (__bf16)f2; v[3] = (__bf16)f3;
    return __builtin_bit_cast(uint2, v);
}

// ---------------------------------------------------------------------------
// Router: rh = relu(X @ rW1 + rb1); logits = rh @ rW2 + rb2; top-2 + renorm.
// One block = 64 samples, 256 threads, fp32 (selection is discrete).
// ---------------------------------------------------------------------------
__global__ __launch_bounds__(256) void router_kernel(
    const float* __restrict__ x, const float* __restrict__ rW1,
    const float* __restrict__ rb1, const float* __restrict__ rW2,
    const float* __restrict__ rb2,
    int* __restrict__ cnt, int* __restrict__ list, float* __restrict__ wlist)
{
    __shared__ float xt[64][33];
    __shared__ float wt[32][RHID];
    __shared__ float rh[64][RHID + 1];

    const int tid = threadIdx.x;
    const int bs  = blockIdx.x * 64;
    const int tx  = tid & 31;
    const int ty  = tid >> 5;

    float acc[8][4];
#pragma unroll
    for (int r = 0; r < 8; ++r)
#pragma unroll
        for (int c = 0; c < 4; ++c) acc[r][c] = 0.f;

    for (int k0 = 0; k0 < DIN; k0 += 32) {
        {
            int r  = tid >> 2;
            int kk = (tid & 3) * 8;
            const float4* src = reinterpret_cast<const float4*>(&x[(size_t)(bs + r) * DIN + k0 + kk]);
            float4 a = src[0], b = src[1];
            xt[r][kk + 0] = a.x; xt[r][kk + 1] = a.y; xt[r][kk + 2] = a.z; xt[r][kk + 3] = a.w;
            xt[r][kk + 4] = b.x; xt[r][kk + 5] = b.y; xt[r][kk + 6] = b.z; xt[r][kk + 7] = b.w;
        }
        {
            int kk  = tid >> 3;
            int col = (tid & 7) * 16;
            const float4* src = reinterpret_cast<const float4*>(&rW1[(size_t)(k0 + kk) * RHID + col]);
#pragma unroll
            for (int q = 0; q < 4; ++q) {
                float4 w4 = src[q];
                wt[kk][col + q * 4 + 0] = w4.x; wt[kk][col + q * 4 + 1] = w4.y;
                wt[kk][col + q * 4 + 2] = w4.z; wt[kk][col + q * 4 + 3] = w4.w;
            }
        }
        __syncthreads();
#pragma unroll 4
        for (int kk = 0; kk < 32; ++kk) {
            float w0 = wt[kk][tx * 4 + 0], w1 = wt[kk][tx * 4 + 1];
            float w2 = wt[kk][tx * 4 + 2], w3 = wt[kk][tx * 4 + 3];
#pragma unroll
            for (int r = 0; r < 8; ++r) {
                float a = xt[ty * 8 + r][kk];
                acc[r][0] = fmaf(a, w0, acc[r][0]);
                acc[r][1] = fmaf(a, w1, acc[r][1]);
                acc[r][2] = fmaf(a, w2, acc[r][2]);
                acc[r][3] = fmaf(a, w3, acc[r][3]);
            }
        }
        __syncthreads();
    }

#pragma unroll
    for (int c = 0; c < 4; ++c) {
        float b = rb1[tx * 4 + c];
#pragma unroll
        for (int r = 0; r < 8; ++r)
            rh[ty * 8 + r][tx * 4 + c] = fmaxf(acc[r][c] + b, 0.f);
    }
    __syncthreads();

    if (tid < 64) {
        float lg[NEXP];
#pragma unroll
        for (int e = 0; e < NEXP; ++e) lg[e] = rb2[e];
        for (int j = 0; j < RHID; ++j) {
            float h = rh[tid][j];
#pragma unroll
            for (int e = 0; e < NEXP; ++e) lg[e] = fmaf(h, rW2[j * NEXP + e], lg[e]);
        }
        int i0 = 0;
#pragma unroll
        for (int e = 1; e < NEXP; ++e) if (lg[e] > lg[i0]) i0 = e;
        int i1 = (i0 == 0) ? 1 : 0;
#pragma unroll
        for (int e = 0; e < NEXP; ++e) if (e != i0 && lg[e] > lg[i1]) i1 = e;
        float w0 = 1.f / (1.f + expf(lg[i1] - lg[i0]));   // softmax over top-2
        float w1 = 1.f - w0;
        int b = bs + tid;
        int p0 = atomicAdd(&cnt[i0], 1);
        list[i0 * NB + p0]  = b;
        wlist[i0 * NB + p0] = w0;
        int p1 = atomicAdd(&cnt[i1], 1);
        list[i1 * NB + p1]  = b;
        wlist[i1 * NB + p1] = w1;
    }
}

// ---------------------------------------------------------------------------
// Expert kernel: per (expert, m-tile, h-tile), 128x64 tile of
// relu(Xsub @ eW1[e] + eb1[e]) via bf16 WMMA. Ping-pong LDS buffers,
// manually 2x-unrolled pipeline, hoisted gather pointers + fragment
// coordinates, packed-cvt b64 fragment stores, weight-stream prefetch.
// Then contract with eW2[e][h-tile,:10], weighted atomic scatter to out.
// ---------------------------------------------------------------------------
__global__ __launch_bounds__(256) void expert_kernel(
    const float* __restrict__ x,
    const float* __restrict__ eW1, const float* __restrict__ eb1,
    const float* __restrict__ eW2, const float* __restrict__ eb2,
    const int* __restrict__ cnt, const int* __restrict__ list,
    const float* __restrict__ wlist, float* __restrict__ out)
{
    const int e  = blockIdx.z;
    const int mt = blockIdx.y;
    const int ht = blockIdx.x;
    const int n  = cnt[e];
    const int m0 = mt * TM;
    if (m0 >= n) return;                 // uniform: EXEC stays full for WMMA
    const int h0 = ht * THD;

    __shared__ __align__(32) unsigned short aF[2][8][32][16]; // 16 KB (8 m-frags)
    __shared__ __align__(32) unsigned short bF[2][4][32][16]; //  8 KB (4 n-frags)
    __shared__ float eh[TM][THD];                             // 32 KB
    __shared__ float w2t[THD][NCLS];
    __shared__ int   smp[TM];
    __shared__ float swt[TM];

    const int tid  = threadIdx.x;
    const int lane = tid & 31;
    const int wave = tid >> 5;
    const int wm   = wave >> 1;   // 0..3 : 32-row stripe
    const int wn   = wave & 1;    // 0..1 : 32-col stripe

    if (tid < TM) {
        int g = m0 + tid;
        if (g < n) { smp[tid] = list[e * NB + g]; swt[tid] = wlist[e * NB + g]; }
        else       { smp[tid] = 0;               swt[tid] = 0.f; }
    }
    __syncthreads();

    // ---- thread-constant staging geometry (hoisted out of the K loop) ----
    const int rbase = tid >> 3;                 // 0..31
    const int kka   = (tid & 7) * 4;            // 0..28
    const int lnA   = (rbase & 15) + (((kka >> 3) & 1) << 4);
    const int jA    = (kka & 7) + (((kka >> 4) & 1) << 3);
    const int mf0   = rbase >> 4;               // 0..1

    const float* aptr[4];
#pragma unroll
    for (int p = 0; p < 4; ++p)
        aptr[p] = x + (size_t)smp[rbase + p * 32] * DIN + kka;

    const int kkB  = (tid >> 5) * 4;            // 0,4,...,28
    const int hhB  = (tid & 31) * 2;            // 0..62
    const int jB   = (kkB & 7) + (((kkB >> 4) & 1) << 3);
    const int nf0  = hhB >> 4;
    const int lnB0 = (hhB & 15) + (((kkB >> 3) & 1) << 4);
    const float* bptr = eW1 + (size_t)e * DIN * EHID + (size_t)kkB * EHID + h0 + hhB;

    auto stage = [&](int s, int k0) {
#pragma unroll
        for (int p = 0; p < 4; ++p) {
            const float4 xv = *reinterpret_cast<const float4*>(aptr[p] + k0);
            *reinterpret_cast<uint2*>(&aF[s][mf0 + 2 * p][lnA][jA]) =
                pack4bf(xv.x, xv.y, xv.z, xv.w);
        }
        const float* wr = bptr + (size_t)k0 * EHID;
        float2 r0 = *reinterpret_cast<const float2*>(wr);
        float2 r1 = *reinterpret_cast<const float2*>(wr + EHID);
        float2 r2 = *reinterpret_cast<const float2*>(wr + 2 * EHID);
        float2 r3 = *reinterpret_cast<const float2*>(wr + 3 * EHID);
        *reinterpret_cast<uint2*>(&bF[s][nf0][lnB0][jB])     = pack4bf(r0.x, r1.x, r2.x, r3.x);
        *reinterpret_cast<uint2*>(&bF[s][nf0][lnB0 + 1][jB]) = pack4bf(r0.y, r1.y, r2.y, r3.y);
        if (k0 + 2 * KC < DIN)
            __builtin_prefetch(wr + (size_t)(2 * KC) * EHID, 0, 0);  // global_prefetch_b8
    };

    v8f c00 = {}, c01 = {}, c10 = {}, c11 = {};

    auto mma = [&](int s) {
        v16bf a0 = *reinterpret_cast<const v16bf*>(&aF[s][wm * 2 + 0][lane][0]);
        v16bf a1 = *reinterpret_cast<const v16bf*>(&aF[s][wm * 2 + 1][lane][0]);
        v16bf b0 = *reinterpret_cast<const v16bf*>(&bF[s][wn * 2 + 0][lane][0]);
        v16bf b1 = *reinterpret_cast<const v16bf*>(&bF[s][wn * 2 + 1][lane][0]);
        c00 = __builtin_amdgcn_wmma_f32_16x16x32_bf16(false, a0, false, b0, (short)0, c00, false, false);
        c01 = __builtin_amdgcn_wmma_f32_16x16x32_bf16(false, a0, false, b1, (short)0, c01, false, false);
        c10 = __builtin_amdgcn_wmma_f32_16x16x32_bf16(false, a1, false, b0, (short)0, c10, false, false);
        c11 = __builtin_amdgcn_wmma_f32_16x16x32_bf16(false, a1, false, b1, (short)0, c11, false, false);
    };

    stage(0, 0);
    __syncthreads();

    for (int it = 0; it < NITER; it += 2) {     // NITER even; 2x unrolled ping-pong
        stage(1, (it + 1) * KC);                // it+1 <= NITER-1 always
        mma(0);
        __syncthreads();
        if (it + 2 < NITER) stage(0, (it + 2) * KC);
        mma(1);
        __syncthreads();
    }

    // ---- bias + relu, spill to LDS (C layout: col = lane&15, row = v + (lane>=16)*8) ----
    {
        const int col = lane & 15;
        const int rhi = (lane >> 4) * 8;
        v8f* accs[4] = {&c00, &c01, &c10, &c11};
#pragma unroll
        for (int f = 0; f < 4; ++f) {
            int i = f >> 1, j = f & 1;
            int hloc = wn * 32 + j * 16 + col;
            float bias = eb1[e * EHID + h0 + hloc];
            v8f acc = *accs[f];
#pragma unroll
            for (int v = 0; v < 8; ++v) {
                float val = fmaxf(acc[v] + bias, 0.f);
                eh[wm * 32 + i * 16 + rhi + v][hloc] = val;
            }
        }
    }
    // ---- stage eW2 tile (64 x 10 fp32) ----
    for (int idx = tid; idx < THD * NCLS; idx += 256) {
        int hh = idx / NCLS, c = idx % NCLS;
        w2t[hh][c] = eW2[(size_t)e * EHID * NCLS + (size_t)(h0 + hh) * NCLS + c];
    }
    __syncthreads();

    // ---- layer-2 contraction + weighted atomic accumulate ----
    const int nloc = min(TM, n - m0);
    for (int idx = tid; idx < TM * NCLS; idx += 256) {
        int r = idx / NCLS, c = idx % NCLS;
        if (r < nloc) {
            float s = 0.f;
#pragma unroll 8
            for (int hh = 0; hh < THD; ++hh) s = fmaf(eh[r][hh], w2t[hh][c], s);
            float wgt = swt[r];
            float contrib = wgt * s;
            if (ht == 0) contrib = fmaf(wgt, eb2[e * NCLS + c], contrib);
            atomicAdd(&out[(size_t)smp[r] * NCLS + c], contrib);
        }
    }
}

extern "C" void kernel_launch(void* const* d_in, const int* in_sizes, int n_in,
                              void* d_out, int out_size, void* d_ws, size_t ws_size,
                              hipStream_t stream) {
    const float* x   = (const float*)d_in[0];
    const float* rW1 = (const float*)d_in[1];
    const float* rb1 = (const float*)d_in[2];
    const float* rW2 = (const float*)d_in[3];
    const float* rb2 = (const float*)d_in[4];
    const float* eW1 = (const float*)d_in[5];
    const float* eb1 = (const float*)d_in[6];
    const float* eW2 = (const float*)d_in[7];
    const float* eb2 = (const float*)d_in[8];
    float* out = (float*)d_out;

    int*   cnt   = (int*)d_ws;
    int*   list  = cnt + 8;                       // 8 * 1024 ints
    float* wlist = (float*)(list + NEXP * NB);    // 8 * 1024 floats

    hipMemsetAsync(cnt, 0, NEXP * sizeof(int), stream);
    hipMemsetAsync(d_out, 0, (size_t)out_size * sizeof(float), stream);

    router_kernel<<<NB / 64, 256, 0, stream>>>(x, rW1, rb1, rW2, rb2, cnt, list, wlist);

    dim3 grid(EHID / THD, NB / TM, NEXP);         // 32 x 8 x 8
    expert_kernel<<<grid, 256, 0, stream>>>(x, eW1, eb1, eW2, eb2, cnt, list, wlist, out);
}
